// ParallelExpert_17111149707607
// MI455X (gfx1250) — matrix operand order repaired and checked by reading, not compiled, not run on hardware
//
#include <hip/hip_runtime.h>
#include <hip/hip_bf16.h>
#include <cstdint>

typedef __attribute__((ext_vector_type(16))) __bf16 v16bf;
typedef __attribute__((ext_vector_type(8)))  float  v8f;
typedef __attribute__((ext_vector_type(4)))  __bf16 v4bf;

#define BM   128
#define BK   32
#define LSTR 40   // padded LDS row stride (elements): 80B, conflict-free & 16B aligned

// ---- CDNA5 async global->LDS path (guarded: falls back to ld/st if absent) ----
#if defined(__has_builtin)
#if __has_builtin(__builtin_amdgcn_global_load_async_to_lds_b128) && \
    __has_builtin(__builtin_amdgcn_s_wait_asynccnt)
#define USE_ASYNC_LDS 1
#endif
#endif
#ifndef USE_ASYNC_LDS
#define USE_ASYNC_LDS 0
#endif

// Builtin prototype (from hipcc diagnostic): param0 = 'int __vector_size__(16) __device__ *'
// i.e. v4i in address_space(1); param1 expected to be v4i in address_space(3).
typedef int v4i_t __attribute__((__vector_size__(16)));
typedef __attribute__((address_space(1))) v4i_t* g4p_t;
typedef __attribute__((address_space(3))) v4i_t* l4p_t;

__device__ __forceinline__ void copy16_g2l(const __bf16* g, __bf16* l) {
#if USE_ASYNC_LDS
    __builtin_amdgcn_global_load_async_to_lds_b128(
        (g4p_t)(uintptr_t)g,
        (l4p_t)(unsigned)(uintptr_t)l,
        0, 0);
#else
    *reinterpret_cast<uint4*>(l) = *reinterpret_cast<const uint4*>(g);
#endif
}

__device__ __forceinline__ void tile_fence() {
#if USE_ASYNC_LDS
    __builtin_amdgcn_s_wait_asynccnt(0);
#endif
}

struct Frag2 { uint4 a, b; };

__device__ inline v16bf ld_frag(const __bf16* p0, const __bf16* p1) {
    Frag2 f;
    f.a = *reinterpret_cast<const uint4*>(p0);
    f.b = *reinterpret_cast<const uint4*>(p1);
    return __builtin_bit_cast(v16bf, f);
}

// ---------------- prepass: fp32 -> bf16 convert (x) ----------------
__global__ __launch_bounds__(256) void cvt_f32_bf16(const float* __restrict__ src,
                                                    __bf16* __restrict__ dst, size_t n4) {
    size_t i = (size_t)blockIdx.x * 256 + threadIdx.x;
    if (i >= n4) return;
    float4 f = reinterpret_cast<const float4*>(src)[i];
    v4bf b;
    b[0] = (__bf16)f.x; b[1] = (__bf16)f.y; b[2] = (__bf16)f.z; b[3] = (__bf16)f.w;
    reinterpret_cast<v4bf*>(dst)[i] = b;
}

// ---------- prepass: per-expert transpose [R][C] fp32 -> [C][R] bf16 ----------
__global__ __launch_bounds__(256) void transpose_to_bf16(const float* __restrict__ src,
                                                         __bf16* __restrict__ dst,
                                                         int R, int C) {
    __shared__ float tile[32][33];
    const int bx = blockIdx.x * 32;           // col base in src
    const int by = blockIdx.y * 32;           // row base in src
    const float* s = src + (size_t)blockIdx.z * R * C;
    __bf16*      d = dst + (size_t)blockIdx.z * R * C;
    const int tx = threadIdx.x & 31;
    const int ty = threadIdx.x >> 5;          // 0..7
#pragma unroll
    for (int i = 0; i < 32; i += 8)
        tile[ty + i][tx] = s[(size_t)(by + ty + i) * C + bx + tx];
    __syncthreads();
#pragma unroll
    for (int i = 0; i < 32; i += 8)
        d[(size_t)(bx + ty + i) * R + by + tx] = (__bf16)tile[tx][ty + i];
}

// ---------------- GEMM1 fused with SwiGLU ----------------
// x_bf [E][T][1024] row-major, w1t [E][8192][1024] (N-major), act [E][T][4096] bf16
__global__ __launch_bounds__(256) void gemm1_swiglu(const __bf16* __restrict__ xbf,
                                                    const __bf16* __restrict__ w1t,
                                                    __bf16* __restrict__ act) {
    constexpr int K = 1024, DHID = 4096, INTER = 8192, T = 2048;
    __shared__ __bf16 sA[2][BM * LSTR];
    __shared__ __bf16 sB[2][BM * LSTR];   // rows 0..63 gate, 64..127 val

    const int e  = blockIdx.z;
    const int m0 = blockIdx.x * BM;
    const int n0 = blockIdx.y * 64;
    const int tid   = threadIdx.x;
    const int lane  = tid & 31;
    const int wid   = tid >> 5;
    const int waveM = wid & 1;     // 0..1 -> 64 rows each
    const int waveN = wid >> 1;    // 0..3 -> 16 cols each
    const int khalf = lane >> 4;
    const int l16   = lane & 15;

    const __bf16* Asrc = xbf + ((size_t)e * T + m0) * K;
    const __bf16* Bg   = w1t + ((size_t)e * INTER + n0) * K;
    const __bf16* Bv   = w1t + ((size_t)e * INTER + DHID + n0) * K;

    v8f accg[4] = {};
    v8f accv[4] = {};

    auto load_tiles = [&](int buf, int kt) {
        const int k0 = kt * BK;
#pragma unroll
        for (int s = 0; s < 2; ++s) {                 // A: 128 rows x 4 chunks
            int slot = tid + s * 256;
            int row = slot >> 2, c8 = (slot & 3) * 8;
            copy16_g2l(Asrc + (size_t)row * K + k0 + c8, &sA[buf][row * LSTR + c8]);
        }
        {                                             // B: 64 gate rows + 64 val rows
            int row = tid >> 2, c8 = (tid & 3) * 8;
            copy16_g2l(Bg + (size_t)row * K + k0 + c8, &sB[buf][row * LSTR + c8]);
            copy16_g2l(Bv + (size_t)row * K + k0 + c8, &sB[buf][(64 + row) * LSTR + c8]);
        }
    };

    load_tiles(0, 0);
    const int nk = K / BK;    // 32
    for (int kt = 0; kt < nk; ++kt) {
        tile_fence();          // this wave's async writes (buffer kt&1) have landed
        __syncthreads();       // everyone done reading buffer (kt+1)&1
        if (kt + 1 < nk) load_tiles((kt + 1) & 1, kt + 1);
        const int buf = kt & 1;

        v16bf a[4];
#pragma unroll
        for (int i = 0; i < 4; ++i) {
            const __bf16* p = &sA[buf][(waveM * 64 + i * 16 + l16) * LSTR + khalf * 16];
            a[i] = ld_frag(p, p + 8);
        }
        const __bf16* pg = &sB[buf][(waveN * 16 + l16) * LSTR + khalf * 16];
        v16bf bg = ld_frag(pg, pg + 8);
        const __bf16* pv = &sB[buf][(64 + waveN * 16 + l16) * LSTR + khalf * 16];
        v16bf bv = ld_frag(pv, pv + 8);

#pragma unroll
        for (int i = 0; i < 4; ++i) {
            accg[i] = __builtin_amdgcn_wmma_f32_16x16x32_bf16(false, a[i], false, bg,
                                                              (short)0, accg[i], false, false);
            accv[i] = __builtin_amdgcn_wmma_f32_16x16x32_bf16(false, a[i], false, bv,
                                                              (short)0, accv[i], false, false);
        }
    }

    // epilogue: act = silu(gate) * val, stored bf16
    const int nG = n0 + waveN * 16 + l16;
    __bf16* actE = act + (size_t)e * T * DHID;
#pragma unroll
    for (int i = 0; i < 4; ++i) {
        const int mBase = m0 + waveM * 64 + i * 16 + khalf * 8;
        __bf16* dst = actE + (size_t)mBase * DHID + nG;
#pragma unroll
        for (int r = 0; r < 8; ++r) {
            float g = accg[i][r];
            float v = accv[i][r];
            float s = (g / (1.0f + __expf(-g))) * v;
            dst[(size_t)r * DHID] = (__bf16)s;
        }
    }
}

// ---------------- GEMM2 ----------------
// act [E][T][4096] bf16, w2t [E][1024][4096] (N-major), out [E][T][1024] fp32
__global__ __launch_bounds__(256) void gemm2_kernel(const __bf16* __restrict__ act,
                                                    const __bf16* __restrict__ w2t,
                                                    float* __restrict__ out) {
    constexpr int K = 4096, DOUT = 1024, T = 2048;
    __shared__ __bf16 sA[2][BM * LSTR];
    __shared__ __bf16 sB[2][BM * LSTR];

    const int e  = blockIdx.z;
    const int m0 = blockIdx.x * BM;
    const int n0 = blockIdx.y * BM;
    const int tid   = threadIdx.x;
    const int lane  = tid & 31;
    const int wid   = tid >> 5;
    const int waveM = wid & 1;     // 64 rows
    const int waveN = wid >> 1;    // 32 cols (2 subtiles)
    const int khalf = lane >> 4;
    const int l16   = lane & 15;

    const __bf16* Asrc = act + ((size_t)e * T + m0) * K;
    const __bf16* Bsrc = w2t + ((size_t)e * DOUT + n0) * K;

    v8f acc[4][2] = {};

    auto load_tiles = [&](int buf, int kt) {
        const int k0 = kt * BK;
#pragma unroll
        for (int s = 0; s < 2; ++s) {
            int slot = tid + s * 256;
            int row = slot >> 2, c8 = (slot & 3) * 8;
            copy16_g2l(Asrc + (size_t)row * K + k0 + c8, &sA[buf][row * LSTR + c8]);
            copy16_g2l(Bsrc + (size_t)row * K + k0 + c8, &sB[buf][row * LSTR + c8]);
        }
    };

    load_tiles(0, 0);
    const int nk = K / BK;    // 128
    for (int kt = 0; kt < nk; ++kt) {
        tile_fence();
        __syncthreads();
        if (kt + 1 < nk) load_tiles((kt + 1) & 1, kt + 1);
        const int buf = kt & 1;

        v16bf a[4];
#pragma unroll
        for (int i = 0; i < 4; ++i) {
            const __bf16* p = &sA[buf][(waveM * 64 + i * 16 + l16) * LSTR + khalf * 16];
            a[i] = ld_frag(p, p + 8);
        }
        v16bf b[2];
#pragma unroll
        for (int j = 0; j < 2; ++j) {
            const __bf16* p = &sB[buf][(waveN * 32 + j * 16 + l16) * LSTR + khalf * 16];
            b[j] = ld_frag(p, p + 8);
        }
#pragma unroll
        for (int i = 0; i < 4; ++i)
#pragma unroll
            for (int j = 0; j < 2; ++j)
                acc[i][j] = __builtin_amdgcn_wmma_f32_16x16x32_bf16(false, a[i], false, b[j],
                                                                    (short)0, acc[i][j], false, false);
    }

    float* outE = out + (size_t)e * T * DOUT;
#pragma unroll
    for (int i = 0; i < 4; ++i) {
        const int mBase = m0 + waveM * 64 + i * 16 + khalf * 8;
#pragma unroll
        for (int j = 0; j < 2; ++j) {
            const int nG = n0 + waveN * 32 + j * 16 + l16;
#pragma unroll
            for (int r = 0; r < 8; ++r)
                outE[(size_t)(mBase + r) * DOUT + nG] = acc[i][j][r];
        }
    }
}

extern "C" void kernel_launch(void* const* d_in, const int* in_sizes, int n_in,
                              void* d_out, int out_size, void* d_ws, size_t ws_size,
                              hipStream_t stream) {
    constexpr int E = 8, T = 2048, DIN = 1024, DHID = 4096, DOUT = 1024, INTER = 2 * DHID;

    const float* x  = (const float*)d_in[0];   // [E][T][DIN]
    const float* w1 = (const float*)d_in[1];   // [E][DIN][INTER]
    const float* w2 = (const float*)d_in[2];   // [E][DHID][DOUT]
    float* out = (float*)d_out;                // [E][T][DOUT]

    char* ws = (char*)d_ws;
    const size_t xbf_bytes = (size_t)E * T * DIN * 2;        //  32 MiB
    const size_t w1t_bytes = (size_t)E * INTER * DIN * 2;    // 128 MiB
    const size_t w2t_bytes = (size_t)E * DOUT * DHID * 2;    //  64 MiB
    __bf16* xbf  = (__bf16*)(ws);
    __bf16* w1t  = (__bf16*)(ws + xbf_bytes);
    __bf16* w2t  = (__bf16*)(ws + xbf_bytes + w1t_bytes);
    __bf16* actb = (__bf16*)(ws + xbf_bytes + w1t_bytes + w2t_bytes);

    // 1) prepass: convert / transpose into bf16 N-major layouts
    size_t nx4 = (size_t)E * T * DIN / 4;
    cvt_f32_bf16<<<dim3((unsigned)(nx4 / 256)), dim3(256), 0, stream>>>(x, xbf, nx4);
    // w1: [DIN][INTER] -> [INTER][DIN]
    transpose_to_bf16<<<dim3(INTER / 32, DIN / 32, E), dim3(256), 0, stream>>>(w1, w1t, DIN, INTER);
    // w2: [DHID][DOUT] -> [DOUT][DHID]
    transpose_to_bf16<<<dim3(DOUT / 32, DHID / 32, E), dim3(256), 0, stream>>>(w2, w2t, DHID, DOUT);

    // 2) GEMM1 + SwiGLU -> act bf16 [E][T][DHID]
    gemm1_swiglu<<<dim3(T / 128, DHID / 64, E), dim3(256), 0, stream>>>(xbf, w1t, actb);

    // 3) GEMM2 -> out fp32 [E][T][DOUT]
    gemm2_kernel<<<dim3(T / 128, DOUT / 128, E), dim3(256), 0, stream>>>(actb, w2t, out);
}
